// QuantLSTMLayer_48515950576198
// MI455X (gfx1250) — compile-verified, matrix operand order checked by reading, not run on hardware
//
#include <hip/hip_runtime.h>
#include <hip/hip_bf16.h>

typedef __bf16 bf16;
typedef __attribute__((ext_vector_type(8)))  bf16  v8bf;
typedef __attribute__((ext_vector_type(16))) bf16  v16bf;
typedef __attribute__((ext_vector_type(8)))  float v8f;

#define T_STEPS 1024
#define BATCH   64
#define IDIM    512
#define HDIM    512
#define GCOLS   (4 * HDIM)          // 2048
#define ROWS    (T_STEPS * BATCH)   // 65536
#define REC_BLOCKS 16

// Assemble a v16bf WMMA operand from two contiguous 16-byte chunks.
// A-matrix 16x32 bf16 lane layout (ISA 7.12.2): lane l (m = l&15, kg = l>>4):
//   halves 0..7  = K = kbase + kg*8 + 0..7
//   halves 8..15 = K = kbase + 16 + kg*8 + 0..7
// B (32x16) mirrors this with lane = column N, so loading row N of a
// [out][in]-major weight matrix gives B = W^T directly.
__device__ inline v16bf load_frag(const bf16* lo, const bf16* hi) {
  v8bf a = *(const v8bf*)lo;
  v8bf b = *(const v8bf*)hi;
  v16bf r;
#pragma unroll
  for (int i = 0; i < 8; ++i) { r[i] = a[i]; r[i + 8] = b[i]; }
  return r;
}

__device__ inline float sigmoidf(float x) { return 1.0f / (1.0f + __expf(-x)); }

// G scratch lives in WMMA D-fragment layout so both producer and consumer use
// coalesced b128 accesses:  G[t][gate][nt][bt][lane][j]  (8 bf16 per lane).
__device__ inline size_t g_frag_off(int t, int g, int nt, int bt, int lane) {
  return ((((size_t)t * 4 + g) * 32 + nt) * 4 + bt) * 256 + (size_t)lane * 8;
}

// ---------------------------------------------------------------- init kernels
__global__ __launch_bounds__(256) void f2bf_kernel(const float* __restrict__ s,
                                                   bf16* __restrict__ d, int n) {
  int i = blockIdx.x * 256 + threadIdx.x;
  if (i < n) d[i] = (bf16)s[i];
}

__global__ __launch_bounds__(256) void bias_kernel(
    const float* __restrict__ bi0, const float* __restrict__ bi1,
    const float* __restrict__ bi2, const float* __restrict__ bi3,
    const float* __restrict__ bh0, const float* __restrict__ bh1,
    const float* __restrict__ bh2, const float* __restrict__ bh3,
    float* __restrict__ o) {
  int i = blockIdx.x * 256 + threadIdx.x;
  if (i < 4 * HDIM) {
    int g = i >> 9, n = i & (HDIM - 1);
    const float* bi = (g == 0) ? bi0 : (g == 1) ? bi1 : (g == 2) ? bi2 : bi3;
    const float* bh = (g == 0) ? bh0 : (g == 1) ? bh1 : (g == 2) ? bh2 : bh3;
    o[i] = bi[n] + bh[n];
  }
}

__global__ __launch_bounds__(256) void init_state(const float* __restrict__ h0,
                                                  bf16* __restrict__ hbuf,
                                                  int* __restrict__ bar) {
  int i = blockIdx.x * 256 + threadIdx.x;
  if (i < BATCH * HDIM) hbuf[i] = (bf16)h0[i];
  if (i < 2) bar[i] = 0;
}

// ------------------------------------------------- phase 1: input projections
// 32x64 macro-tile per wave: 2 M-tiles x 4 N-tiles (same gate), 8 accumulators.
// Each A fragment feeds 4 WMMAs, each B fragment feeds 2 -> 2.7x less L2
// traffic per FLOP than 1 tile/wave.
__global__ __launch_bounds__(256) void lstm_xgemm(const bf16* __restrict__ xb,
                                                  const bf16* __restrict__ wib,
                                                  bf16* __restrict__ G) {
  const int wave = (blockIdx.x * 256 + threadIdx.x) >> 5;
  const int lane = threadIdx.x & 31;
  const int rm   = wave >> 5;        // row macro-tile 0..2047 (32 rows each)
  const int cm   = wave & 31;        // col macro-tile 0..31   (64 cols each)
  const int g    = cm >> 3;          // gate
  const int n0   = (cm & 7) * 64;    // col offset within gate
  const int m    = lane & 15;
  const int kg   = lane >> 4;

  const bf16* A[2];
  const bf16* Bw[4];
#pragma unroll
  for (int mt = 0; mt < 2; ++mt)
    A[mt] = xb + (size_t)(rm * 32 + mt * 16 + m) * IDIM;
#pragma unroll
  for (int nt = 0; nt < 4; ++nt)
    Bw[nt] = wib + (size_t)g * HDIM * IDIM + (size_t)(n0 + nt * 16 + m) * IDIM;

  v8f acc[2][4] = {};
  for (int kb = 0; kb < IDIM / 32; ++kb) {
    const int k0 = kb * 32 + kg * 8;
    __builtin_prefetch(A[0] + k0 + 64, 0, 1);   // global_prefetch_b8
    __builtin_prefetch(A[1] + k0 + 64, 0, 1);
    v16bf af[2], bf_[4];
#pragma unroll
    for (int mt = 0; mt < 2; ++mt) af[mt] = load_frag(A[mt] + k0, A[mt] + k0 + 16);
#pragma unroll
    for (int nt = 0; nt < 4; ++nt) bf_[nt] = load_frag(Bw[nt] + k0, Bw[nt] + k0 + 16);
#pragma unroll
    for (int mt = 0; mt < 2; ++mt)
#pragma unroll
      for (int nt = 0; nt < 4; ++nt)
        acc[mt][nt] = __builtin_amdgcn_wmma_f32_16x16x32_bf16(
            false, af[mt], false, bf_[nt], (short)0, acc[mt][nt], false, false);
  }

  // store in fragment layout: one coalesced b128 store per tile per lane
#pragma unroll
  for (int mt = 0; mt < 2; ++mt) {
    const int rt = rm * 2 + mt;          // global 16-row tile index
    const int t  = rt >> 2;              // 64 rows (= one timestep) per 4 tiles
    const int bt = rt & 3;
#pragma unroll
    for (int nt = 0; nt < 4; ++nt) {
      const int ntg = (cm & 7) * 4 + nt; // n-tile index within gate, 0..31
      v8bf o;
#pragma unroll
      for (int j = 0; j < 8; ++j) o[j] = (bf16)acc[mt][nt][j];
      *(v8bf*)(G + g_frag_off(t, g, ntg, bt, lane)) = o;
    }
  }
}

// ---------------------------------------------------- phase 2: the recurrence
// 16 blocks x 8 waves = 128 waves; wave w owns (b,n) tile (w>>5, w&31) and
// computes all 4 gates for it. c lives in VGPRs for all 1024 steps.
__global__ __launch_bounds__(256) void lstm_rec(
    const bf16* __restrict__ G, const bf16* __restrict__ whb,
    const float* __restrict__ bias, bf16* __restrict__ hbuf,
    const float* __restrict__ c0, float* __restrict__ out,
    int* __restrict__ bar) {
  const int wid  = blockIdx.x * 8 + (threadIdx.x >> 5);  // 0..127
  const int lane = threadIdx.x & 31;
  const int bt   = wid >> 5;       // 0..3   batch tile
  const int nt   = wid & 31;       // 0..31  hidden tile
  const int m    = lane & 15;
  const int kg   = lane >> 4;
  const int n    = nt * 16 + m;

  float bg[4], creg[8], hlast[8];
#pragma unroll
  for (int g = 0; g < 4; ++g) bg[g] = bias[g * HDIM + n];
#pragma unroll
  for (int j = 0; j < 8; ++j)
    creg[j] = c0[(size_t)(bt * 16 + j + 8 * kg) * HDIM + n];

  const bf16* Bw[4];
#pragma unroll
  for (int g = 0; g < 4; ++g)
    Bw[g] = whb + (size_t)g * HDIM * HDIM + (size_t)n * HDIM;

  int* cnt = bar;
  int* gen = bar + 1;

  for (int t = 0; t < T_STEPS; ++t) {
    const bf16* hp = hbuf + (size_t)(t & 1) * (BATCH * HDIM)
                          + (size_t)(bt * 16 + m) * HDIM;
    // accumulators start at precomputed input-path gates (+ bias); one
    // coalesced b128 load per gate thanks to the fragment layout of G
    v8f acc[4];
#pragma unroll
    for (int g = 0; g < 4; ++g) {
      v8bf gv = *(const v8bf*)(G + g_frag_off(t, g, nt, bt, lane));
#pragma unroll
      for (int j = 0; j < 8; ++j) acc[g][j] = (float)gv[j] + bg[g];
    }
    // h @ W_h^T for all 4 gates; A fragment shared across gates
    for (int kb = 0; kb < HDIM / 32; ++kb) {
      const int k0 = kb * 32 + kg * 8;
      v16bf av = load_frag(hp + k0, hp + k0 + 16);
#pragma unroll
      for (int g = 0; g < 4; ++g) {
        v16bf bv = load_frag(Bw[g] + k0, Bw[g] + k0 + 16);
        acc[g] = __builtin_amdgcn_wmma_f32_16x16x32_bf16(
            false, av, false, bv, (short)0, acc[g], false, false);
      }
    }
    // elementwise gate math; write h (f32 -> out, bf16 -> other h buffer)
    bf16* hq = hbuf + (size_t)((t + 1) & 1) * (BATCH * HDIM);
#pragma unroll
    for (int j = 0; j < 8; ++j) {
      const int b = bt * 16 + j + 8 * kg;
      float ig = sigmoidf(acc[0][j]);
      float fg = sigmoidf(acc[1][j]);
      float ag = tanhf(acc[2][j]);
      float og = sigmoidf(acc[3][j]);
      float cv = fg * creg[j] + ig * ag;
      creg[j]  = cv;
      float h  = og * tanhf(cv);
      hlast[j] = h;
      out[((size_t)t * BATCH + b) * HDIM + n] = h;
      hq[(size_t)b * HDIM + n] = (bf16)h;
    }
    // device-wide sense-reversing barrier (16 resident blocks)
    __threadfence();
    __syncthreads();
    if (threadIdx.x == 0) {
      int g0 = __hip_atomic_load(gen, __ATOMIC_ACQUIRE, __HIP_MEMORY_SCOPE_AGENT);
      if (__hip_atomic_fetch_add(cnt, 1, __ATOMIC_ACQ_REL,
                                 __HIP_MEMORY_SCOPE_AGENT) == REC_BLOCKS - 1) {
        __hip_atomic_store(cnt, 0, __ATOMIC_RELAXED, __HIP_MEMORY_SCOPE_AGENT);
        __hip_atomic_fetch_add(gen, 1, __ATOMIC_ACQ_REL,
                               __HIP_MEMORY_SCOPE_AGENT);
      } else {
        while (__hip_atomic_load(gen, __ATOMIC_ACQUIRE,
                                 __HIP_MEMORY_SCOPE_AGENT) == g0)
          __builtin_amdgcn_s_sleep(1);
      }
    }
    __syncthreads();
  }

  // hT and cT appended after outputs
  float* hT = out + (size_t)T_STEPS * BATCH * HDIM;
  float* cT = hT + BATCH * HDIM;
#pragma unroll
  for (int j = 0; j < 8; ++j) {
    const int b = bt * 16 + j + 8 * kg;
    hT[(size_t)b * HDIM + n] = hlast[j];
    cT[(size_t)b * HDIM + n] = creg[j];
  }
}

// ------------------------------------------------------------------- launcher
extern "C" void kernel_launch(void* const* d_in, const int* in_sizes, int n_in,
                              void* d_out, int out_size, void* d_ws, size_t ws_size,
                              hipStream_t stream) {
  const float* x  = (const float*)d_in[0];
  const float* h0 = (const float*)d_in[1];
  const float* c0 = (const float*)d_in[2];
  const float* wi[4] = {(const float*)d_in[3], (const float*)d_in[4],
                        (const float*)d_in[5], (const float*)d_in[6]};
  const float* wh[4] = {(const float*)d_in[7], (const float*)d_in[8],
                        (const float*)d_in[9], (const float*)d_in[10]};
  float* out = (float*)d_out;

  // workspace carve-up
  char* ws = (char*)d_ws;
  size_t off = 0;
  int* bar = (int*)(ws + off);              off += 256;
  bf16* xb   = (bf16*)(ws + off);           off += (size_t)ROWS * IDIM * 2;      // 64 MB
  bf16* wib  = (bf16*)(ws + off);           off += (size_t)4 * HDIM * IDIM * 2;  // 2 MB
  bf16* whb  = (bf16*)(ws + off);           off += (size_t)4 * HDIM * HDIM * 2;  // 2 MB
  float* bias = (float*)(ws + off);         off += (size_t)4 * HDIM * 4;         // 8 KB
  bf16* hbuf = (bf16*)(ws + off);           off += (size_t)2 * BATCH * HDIM * 2; // 128 KB
  bf16* G    = (bf16*)(ws + off);           off += (size_t)ROWS * GCOLS * 2;     // 256 MB
  (void)off; (void)ws_size; (void)in_sizes; (void)n_in; (void)out_size;

  // phase 0: precision conversion + bias folding + state init
  f2bf_kernel<<<(ROWS * IDIM + 255) / 256, 256, 0, stream>>>(x, xb, ROWS * IDIM);
  for (int g = 0; g < 4; ++g)
    f2bf_kernel<<<(HDIM * IDIM + 255) / 256, 256, 0, stream>>>(
        wi[g], wib + (size_t)g * HDIM * IDIM, HDIM * IDIM);
  for (int g = 0; g < 4; ++g)
    f2bf_kernel<<<(HDIM * HDIM + 255) / 256, 256, 0, stream>>>(
        wh[g], whb + (size_t)g * HDIM * HDIM, HDIM * HDIM);
  bias_kernel<<<(4 * HDIM + 255) / 256, 256, 0, stream>>>(
      (const float*)d_in[11], (const float*)d_in[12], (const float*)d_in[13],
      (const float*)d_in[14], (const float*)d_in[15], (const float*)d_in[16],
      (const float*)d_in[17], (const float*)d_in[18], bias);
  init_state<<<(BATCH * HDIM + 255) / 256, 256, 0, stream>>>(h0, hbuf, bar);

  // phase 1: blocked WMMA GEMM (65536 waves, 32x64 macro-tile each)
  lstm_xgemm<<<(ROWS / 32) * (GCOLS / 64) / 8, 256, 0, stream>>>(xb, wib, G);

  // phase 2: sequential recurrence across 16 cooperating workgroups
  lstm_rec<<<REC_BLOCKS, 256, 0, stream>>>(G, whb, bias, hbuf, c0, out, bar);
}
